// WHISPER_ENCODER_5205500362789
// MI455X (gfx1250) — matrix operand order repaired and checked by reading, not compiled, not run on hardware
//
#include <hip/hip_runtime.h>
#include <hip/hip_bf16.h>

typedef __attribute__((ext_vector_type(16))) _Float16 v16h;
typedef __attribute__((ext_vector_type(8)))  _Float16 v8h;
typedef __attribute__((ext_vector_type(8)))  float    v8f;

#define DEV __device__ __forceinline__

static constexpr int NSAMP   = 479840;
static constexpr int NPADLEN = 480240;   // NSAMP + 400 (center pad 200 each side)
static constexpr int NFFT    = 400;
static constexpr int HOP     = 160;
static constexpr int NFRAMES = 3000;
static constexpr int NBINS   = 201;
static constexpr int NMELS   = 80;
static constexpr int TENC    = 1500;
static constexpr int TP      = 1536;     // padded to multiple of 64 (GEMM block tile)
static constexpr int DD      = 512;
static constexpr int FFD     = 2048;
static constexpr int HH      = 8;
static constexpr int HDIM    = 64;
static constexpr int LEN     = 6;
static constexpr float INV16 = 1.0f / 32768.0f;

DEV float gelu_f(float x) { return 0.5f * x * (1.0f + erff(x * 0.70710678118654752f)); }

DEV unsigned enc_f(float f) {
  unsigned u = __float_as_uint(f);
  return (u & 0x80000000u) ? ~u : (u | 0x80000000u);
}
DEV float dec_f(unsigned e) {
  unsigned u = (e & 0x80000000u) ? (e & 0x7FFFFFFFu) : ~e;
  return __uint_as_float(u);
}

// ---------------------------------------------------------------- front end
__global__ __launch_bounds__(256) void wk_sum(const float* __restrict__ a, int n,
                                              float* __restrict__ acc) {
  __shared__ float red[256];
  int t = threadIdx.x;
  float s = 0.f;
  for (int i = blockIdx.x * 256 + t; i < n; i += gridDim.x * 256) s += a[i];
  red[t] = s; __syncthreads();
  for (int o = 128; o > 0; o >>= 1) { if (t < o) red[t] += red[t + o]; __syncthreads(); }
  if (t == 0) atomicAdd(acc, red[0]);
}

__global__ __launch_bounds__(256) void wk_prep(const float* __restrict__ audio,
                                               const float* __restrict__ accsum,
                                               float* __restrict__ xp) {
  int i = blockIdx.x * 256 + threadIdx.x;
  if (i >= NPADLEN) return;
  float mean = accsum[0] * INV16 / (float)NSAMP;
  if (i < 200 || i >= 200 + NSAMP) { xp[i] = 0.f; return; }
  int j = i - 200;
  float s = audio[j] * INV16 - mean;
  float v = s;
  if (j > 0) { float sp = audio[j - 1] * INV16 - mean; v = s - 0.97f * sp; }
  xp[i] = v;
}

__global__ __launch_bounds__(256) void wk_trig(const float* __restrict__ window,
                                               float* __restrict__ tc, float* __restrict__ ts) {
  int i = blockIdx.x * 256 + threadIdx.x;
  if (i >= NBINS * NFFT) return;
  int k = i / NFFT, n = i % NFFT;
  int r = (k * n) % NFFT;                      // exact periodic reduction
  float ang = 6.283185307179586f * (float)r / (float)NFFT;
  float w = window[n];
  tc[i] = w * cosf(ang);
  ts[i] = w * sinf(ang);
}

__global__ __launch_bounds__(256) void wk_power(const float* __restrict__ xp,
                                                const float* __restrict__ tc,
                                                const float* __restrict__ ts,
                                                float* __restrict__ power) {
  __shared__ float fr[NFFT];
  int f = blockIdx.x, t = threadIdx.x;
  for (int i = t; i < NFFT; i += 256) fr[i] = xp[f * HOP + i];
  __syncthreads();
  for (int k = t; k < NBINS; k += 256) {
    const float* ck = tc + k * NFFT;
    const float* sk = ts + k * NFFT;
    float re = 0.f, im = 0.f;
    for (int n = 0; n < NFFT; ++n) { float x = fr[n]; re += x * ck[n]; im += x * sk[n]; }
    power[(size_t)k * NFRAMES + f] = re * re + im * im;
  }
}

__global__ __launch_bounds__(256) void wk_mel(const float* __restrict__ fb,
                                              const float* __restrict__ power,
                                              float* __restrict__ melraw,
                                              unsigned* __restrict__ gmax) {
  __shared__ unsigned red[256];
  int i = blockIdx.x * 256 + threadIdx.x;
  int t = threadIdx.x;
  unsigned e = 0u;
  if (i < NMELS * NFRAMES) {
    int m = i / NFRAMES, tt = i % NFRAMES;
    float acc = 0.f;
    for (int k = 0; k < NBINS; ++k) acc += fb[m * NBINS + k] * power[(size_t)k * NFRAMES + tt];
    float v = log10f(fmaxf(acc, 1e-7f));
    melraw[i] = v;
    e = enc_f(v);
  }
  red[t] = e; __syncthreads();
  for (int o = 128; o > 0; o >>= 1) { if (t < o) red[t] = max(red[t], red[t + o]); __syncthreads(); }
  if (t == 0) atomicMax(gmax, red[0]);
}

__global__ __launch_bounds__(256) void wk_melnorm(const float* __restrict__ melraw,
                                                  const unsigned* __restrict__ gmax,
                                                  float* __restrict__ mel) {
  int i = blockIdx.x * 256 + threadIdx.x;
  if (i >= NMELS * NFRAMES) return;
  float g = dec_f(*gmax);
  float v = fmaxf(melraw[i], g - 8.f);
  mel[i] = (v + 4.f) * 0.25f;
}

__global__ __launch_bounds__(256) void wk_conv1(const float* __restrict__ mel,
                                                const float* __restrict__ w,
                                                const float* __restrict__ b,
                                                float* __restrict__ h1) {
  int i = blockIdx.x * 256 + threadIdx.x;
  if (i >= DD * NFRAMES) return;
  int o = i / NFRAMES, t = i % NFRAMES;
  float acc = b[o];
  for (int c = 0; c < NMELS; ++c) {
    const float* wr = w + ((size_t)o * NMELS + c) * 3;
    const float* mr = mel + (size_t)c * NFRAMES;
    for (int k = 0; k < 3; ++k) {
      int tt = t + k - 1;
      if (tt >= 0 && tt < NFRAMES) acc += wr[k] * mr[tt];
    }
  }
  h1[i] = gelu_f(acc);
}

__global__ __launch_bounds__(256) void wk_conv2(const float* __restrict__ h1,
                                                const float* __restrict__ w,
                                                const float* __restrict__ b,
                                                const float* __restrict__ pos,
                                                float* __restrict__ x,
                                                _Float16* __restrict__ xh) {
  int i = blockIdx.x * 256 + threadIdx.x;
  if (i >= DD * TENC) return;
  int o = i / TENC, t = i % TENC;
  float acc = b[o];
  for (int c = 0; c < DD; ++c) {
    const float* wr = w + ((size_t)o * DD + c) * 3;
    const float* hr = h1 + (size_t)c * NFRAMES;
    for (int k = 0; k < 3; ++k) {
      int tt = 2 * t + k - 1;
      if (tt >= 0 && tt < NFRAMES) acc += wr[k] * hr[tt];
    }
  }
  float g = gelu_f(acc) + pos[(size_t)t * DD + o];
  x[(size_t)t * DD + o]  = g;
  xh[(size_t)t * DD + o] = (_Float16)g;
}

// ---------------------------------------------------------------- layernorm
__global__ __launch_bounds__(128) void wk_ln(const float* __restrict__ X,
                                             const float* __restrict__ w,
                                             const float* __restrict__ b,
                                             _Float16* __restrict__ O) {
  __shared__ float red[128];
  int row = blockIdx.x, t = threadIdx.x;
  const float* xr = X + (size_t)row * DD;
  float s = 0.f;
  for (int i = t; i < DD; i += 128) s += xr[i];
  red[t] = s; __syncthreads();
  for (int o = 64; o > 0; o >>= 1) { if (t < o) red[t] += red[t + o]; __syncthreads(); }
  float mean = red[0] / (float)DD; __syncthreads();
  float s2 = 0.f;
  for (int i = t; i < DD; i += 128) { float d = xr[i] - mean; s2 += d * d; }
  red[t] = s2; __syncthreads();
  for (int o = 64; o > 0; o >>= 1) { if (t < o) red[t] += red[t + o]; __syncthreads(); }
  float rstd = rsqrtf(red[0] / (float)DD + 1e-5f);
  for (int i = t; i < DD; i += 128)
    O[(size_t)row * DD + i] = (_Float16)((xr[i] - mean) * rstd * w[i] + b[i]);
}

// ---------------------------------------------------------------- softmax
__global__ __launch_bounds__(256) void wk_softmax(const float* __restrict__ S,
                                                  _Float16* __restrict__ P) {
  __shared__ float red[256];
  int row = blockIdx.x, t = threadIdx.x;
  const float* sr = S + (size_t)row * TP;
  float mx = -3.4e38f;
  for (int i = t; i < TENC; i += 256) mx = fmaxf(mx, sr[i]);
  red[t] = mx; __syncthreads();
  for (int o = 128; o > 0; o >>= 1) { if (t < o) red[t] = fmaxf(red[t], red[t + o]); __syncthreads(); }
  float m = red[0]; __syncthreads();
  float sum = 0.f;
  for (int i = t; i < TENC; i += 256) sum += expf(sr[i] - m);
  red[t] = sum; __syncthreads();
  for (int o = 128; o > 0; o >>= 1) { if (t < o) red[t] += red[t + o]; __syncthreads(); }
  float inv = 1.f / red[0];
  for (int i = t; i < TENC; i += 256)
    P[(size_t)row * TP + i] = (_Float16)(expf(sr[i] - m) * inv);
  for (int i = TENC + t; i < TP; i += 256)
    P[(size_t)row * TP + i] = (_Float16)0.f;   // keep pad cols zero
}

// ---------------------------------------------------------------- weight packs (f32 -> f16, NT form [N][K])
__global__ __launch_bounds__(256) void wk_pack_qkv(const float* __restrict__ w,  // [H][D][HD]
                                                   _Float16* __restrict__ o) {   // [H*HD][D]
  int i = blockIdx.x * 256 + threadIdx.x;
  if (i >= DD * DD) return;
  int n = i >> 9, d = i & 511;              // n = h*64+e
  int h = n >> 6, e = n & 63;
  o[i] = (_Float16)w[((size_t)h * DD + d) * HDIM + e];
}

__global__ __launch_bounds__(256) void wk_pack_o(const float* __restrict__ w,    // [H][HD][D]
                                                 _Float16* __restrict__ o) {     // [D][H*HD]
  int i = blockIdx.x * 256 + threadIdx.x;
  if (i >= DD * DD) return;
  int d = i >> 9, c = i & 511;
  int h = c >> 6, e = c & 63;
  o[i] = (_Float16)w[((size_t)h * HDIM + e) * DD + d];
}

__global__ __launch_bounds__(256) void wk_pack_nt(const float* __restrict__ w,   // [K][N]
                                                  _Float16* __restrict__ o,      // [N][K]
                                                  int K, int N) {
  int i = blockIdx.x * 256 + threadIdx.x;
  if (i >= K * N) return;
  int n = i / K, k = i % K;
  o[i] = (_Float16)w[(size_t)k * N + n];
}

// ---------------------------------------------------------------- WMMA GEMM
// C[M,N] = A[M,K] * B  (+bias +gelu +residual), A f16 row-major (lda).
// BNT=1: B is [N][K] row-major (ldb). BNT=0: B is [K][N] row-major (ldb).
// M,N multiples of 64; K multiple of 32.
// Block: 256 threads = 8 waves; tile 64(M) x 64(N); each wave computes a
// 32x16 C strip (two 16x16 accumulators, one B fragment reused -> 2 WMMA/step).
// Row-major tiles are staged with CDNA5 direct global->LDS async copies
// (GLOBAL_LOAD_ASYNC_TO_LDS_B128, tracked by ASYNCcnt): no VGPR bounce.
template <int BNT>
__global__ __launch_bounds__(256) void wk_gemm(const _Float16* __restrict__ A, int lda,
                                               const _Float16* __restrict__ B, int ldb,
                                               const float* __restrict__ bias,
                                               const float* __restrict__ resid, int ldr,
                                               float* __restrict__ C32, int ldc32,
                                               _Float16* __restrict__ C16, int ldc16,
                                               int M, int N, int K, int act) {
  __shared__ __align__(16) _Float16 lA[64][40];
  __shared__ __align__(16) _Float16 lB[64][40];   // stored as [n][k]

  const int t = threadIdx.x;
  const int m0 = blockIdx.y * 64;
  const int n0 = blockIdx.x * 64;
  const int lane = t & 31;
  const int wav  = t >> 5;                 // 0..7
  const int wm = (wav >> 2) << 5;          // 0 or 32   (M strip base within tile)
  const int wn = (wav & 3) << 4;           // 0,16,32,48 (N base within tile)
  const int hl  = lane >> 4;               // 0/1 (lanes 0-15 vs 16-31)
  const int l16 = lane & 15;

  const int arow = t >> 2;                 // 0..63
  const int acol = (t & 3) << 3;           // 0,8,16,24
  const int brow = t >> 3;                 // 0..31  (NN staging)
  const int bcol = (t & 7) << 3;           // 0..56  (NN staging)

  const unsigned ldsA = (unsigned)(uintptr_t)&lA[arow][acol];
  const unsigned ldsB = (unsigned)(uintptr_t)&lB[arow][acol];

  v8f acc0 = {};
  v8f acc1 = {};

  for (int kt = 0; kt < K; kt += 32) {
    // stage A tile 64x32 (row-major) via async global->LDS B128 (one 16B chunk/lane)
    {
      const _Float16* ga = A + (size_t)(m0 + arow) * lda + kt + acol;
      asm volatile("global_load_async_to_lds_b128 %0, %1, off"
                   :: "v"(ldsA), "v"(ga) : "memory");
    }
    if (BNT) {
      const _Float16* gb = B + (size_t)(n0 + arow) * ldb + kt + acol;
      asm volatile("global_load_async_to_lds_b128 %0, %1, off"
                   :: "v"(ldsB), "v"(gb) : "memory");
    } else {
      // transposed staging: must bounce through VGPRs
      v8h bv = *(const v8h*)(B + (size_t)(kt + brow) * ldb + n0 + bcol);
#pragma unroll
      for (int i = 0; i < 8; ++i) lB[bcol + i][brow] = bv[i];
    }
    asm volatile("s_wait_asynccnt 0x0" ::: "memory");
    __syncthreads();

    // B fragment (shared by both WMMAs): lane col N=wn+l16; K = hl*8..+7 and 16+hl*8..+7
    v8h blo = *(const v8h*)&lB[wn + l16][hl << 3];
    v8h bhi = *(const v8h*)&lB[wn + l16][16 + (hl << 3)];
    // A fragments for rows wm+l16 and wm+16+l16; K = hl*8..+7 and 16+hl*8..+7
    v8h a0lo = *(const v8h*)&lA[wm + l16][hl << 3];
    v8h a0hi = *(const v8h*)&lA[wm + l16][16 + (hl << 3)];
    v8h a1lo = *(const v8h*)&lA[wm + 16 + l16][hl << 3];
    v8h a1hi = *(const v8h*)&lA[wm + 16 + l16][16 + (hl << 3)];

    v16h bvv, av0, av1;
#pragma unroll
    for (int i = 0; i < 8; ++i) {
      bvv[i] = blo[i]; bvv[i + 8] = bhi[i];
      av0[i] = a0lo[i]; av0[i + 8] = a0hi[i];
      av1[i] = a1lo[i]; av1[i + 8] = a1hi[i];
    }
    acc0 = __builtin_amdgcn_wmma_f32_16x16x32_f16(false, av0, false, bvv,
                                                  (short)0, acc0, false, false);
    acc1 = __builtin_amdgcn_wmma_f32_16x16x32_f16(false, av1, false, bvv,
                                                  (short)0, acc1, false, false);
    __syncthreads();
  }

  const int gn = n0 + wn + l16;
  float bv = bias ? bias[gn] : 0.f;
#pragma unroll
  for (int s = 0; s < 2; ++s) {
    const int gmb = m0 + wm + s * 16 + (hl << 3);
#pragma unroll
    for (int r = 0; r < 8; ++r) {
      float v = (s == 0 ? acc0[r] : acc1[r]) + bv;
      if (act == 1) v = gelu_f(v);
      int gm = gmb + r;
      if (resid) v += resid[(size_t)gm * ldr + gn];
      if (C32) C32[(size_t)gm * ldc32 + gn] = v;
      if (C16) C16[(size_t)gm * ldc16 + gn] = (_Float16)v;
    }
  }
}

// ---------------------------------------------------------------- output rearrange
__global__ __launch_bounds__(256) void wk_keys(const float* __restrict__ Kc,
                                               float* __restrict__ out, size_t base) {
  int i = blockIdx.x * 256 + threadIdx.x;
  if (i >= DD * TENC) return;
  int ne = i / TENC, t = i % TENC;     // keys[l][h][e][t] -> flat (h*64+e)*1500 + t
  out[base + i] = Kc[(size_t)t * DD + ne];
}

__global__ __launch_bounds__(256) void wk_vals(const float* __restrict__ Vc,
                                               float* __restrict__ out, size_t base) {
  int i = blockIdx.x * 256 + threadIdx.x;
  if (i >= DD * TENC) return;
  int t = i / DD, ne = i % DD;
  int h = ne >> 6, e = ne & 63;        // vals[l][h][t][e]
  out[base + (((size_t)h * TENC + t) * HDIM + e)] = Vc[(size_t)t * DD + ne];
}

// ---------------------------------------------------------------- host
extern "C" void kernel_launch(void* const* d_in, const int* in_sizes, int n_in,
                              void* d_out, int out_size, void* d_ws, size_t ws_size,
                              hipStream_t stream) {
  (void)in_sizes; (void)n_in; (void)out_size;
  const float* audio   = (const float*)d_in[0];
  const float* fbank   = (const float*)d_in[1];
  const float* window  = (const float*)d_in[2];
  const float* conv1_w = (const float*)d_in[3];
  const float* conv1_b = (const float*)d_in[4];
  const float* conv2_w = (const float*)d_in[5];
  const float* conv2_b = (const float*)d_in[6];
  const float* pos_emb = (const float*)d_in[7];
  const float* ln1_w   = (const float*)d_in[8];
  const float* ln1_b   = (const float*)d_in[9];
  const float* qw      = (const float*)d_in[10];
  const float* qb      = (const float*)d_in[11];
  const float* kw      = (const float*)d_in[12];
  const float* vw      = (const float*)d_in[13];
  const float* vb      = (const float*)d_in[14];
  const float* ow      = (const float*)d_in[15];
  const float* ob      = (const float*)d_in[16];
  const float* ln2_w   = (const float*)d_in[17];
  const float* ln2_b   = (const float*)d_in[18];
  const float* fc1_w   = (const float*)d_in[19];
  const float* fc1_b   = (const float*)d_in[20];
  const float* fc2_w   = (const float*)d_in[21];
  const float* fc2_b   = (const float*)d_in[22];
  const float* lnf_w   = (const float*)d_in[23];
  const float* lnf_b   = (const float*)d_in[24];
  const float* dec_kw  = (const float*)d_in[25];
  const float* dec_vw  = (const float*)d_in[26];
  const float* dec_vb  = (const float*)d_in[27];
  float* out = (float*)d_out;

  char* wsb = (char*)d_ws;
  size_t cur = 0;
  auto alloc = [&](size_t nbytes) -> size_t {
    size_t o = cur;
    cur += (nbytes + 255) & ~(size_t)255;
    return o;
  };

  size_t o_scal  = alloc(256);
  size_t o_xp    = alloc((size_t)NPADLEN * 4);
  size_t o_tc    = alloc((size_t)NBINS * NFFT * 4);
  size_t o_ts    = alloc((size_t)NBINS * NFFT * 4);
  size_t o_pow   = alloc((size_t)NBINS * NFRAMES * 4);
  size_t o_mraw  = alloc((size_t)NMELS * NFRAMES * 4);
  size_t o_mel   = alloc((size_t)NMELS * NFRAMES * 4);
  size_t o_h1    = alloc((size_t)DD * NFRAMES * 4);
  size_t o_x     = alloc((size_t)TP * DD * 4);
  size_t o_xh    = alloc((size_t)TP * DD * 2);
  size_t o_xn    = alloc((size_t)TP * DD * 2);
  size_t o_q     = alloc((size_t)TP * DD * 2);
  size_t o_k     = alloc((size_t)TP * DD * 2);
  size_t o_v     = alloc((size_t)TP * DD * 2);
  size_t o_ao    = alloc((size_t)TP * DD * 2);
  size_t o_ff    = alloc((size_t)TP * FFD * 2);
  size_t o_S     = alloc((size_t)TP * TP * 4);
  size_t o_P     = alloc((size_t)TP * TP * 2);
  size_t o_wB    = alloc((size_t)FFD * DD * 2);
  size_t o_Kc    = alloc((size_t)TP * DD * 4);
  size_t o_Vc    = alloc((size_t)TP * DD * 4);
  size_t total = cur;

  float*     wsum  = (float*)(wsb + o_scal);
  unsigned*  wmax  = (unsigned*)(wsb + o_scal + 4);
  float*     xp    = (float*)(wsb + o_xp);
  float*     tc    = (float*)(wsb + o_tc);
  float*     ts    = (float*)(wsb + o_ts);
  float*     powr  = (float*)(wsb + o_pow);
  float*     mraw  = (float*)(wsb + o_mraw);
  float*     mel   = (float*)(wsb + o_mel);
  float*     h1    = (float*)(wsb + o_h1);
  float*     x     = (float*)(wsb + o_x);
  _Float16*  xh    = (_Float16*)(wsb + o_xh);
  _Float16*  xn    = (_Float16*)(wsb + o_xn);
  _Float16*  q_h   = (_Float16*)(wsb + o_q);
  _Float16*  k_h   = (_Float16*)(wsb + o_k);
  _Float16*  v_h   = (_Float16*)(wsb + o_v);
  _Float16*  ao_h  = (_Float16*)(wsb + o_ao);
  _Float16*  ff_h  = (_Float16*)(wsb + o_ff);
  float*     S     = (float*)(wsb + o_S);
  _Float16*  P     = (_Float16*)(wsb + o_P);
  _Float16*  wB    = (_Float16*)(wsb + o_wB);
  float*     Kc    = (float*)(wsb + o_Kc);
  float*     Vc    = (float*)(wsb + o_Vc);

  size_t mset = total <= ws_size ? total : ws_size;
  hipMemsetAsync(d_ws, 0, mset, stream);

  auto g1 = [](int n) { return (unsigned)((n + 255) / 256); };
  auto gemmNT = [&](const _Float16* A, int lda, const _Float16* B, int ldb,
                    const float* bias, const float* resid, int ldr,
                    float* c32, int ldc32, _Float16* c16, int ldc16,
                    int M, int N, int K, int act) {
    dim3 grid((unsigned)(N / 64), (unsigned)(M / 64));
    wk_gemm<1><<<grid, 256, 0, stream>>>(A, lda, B, ldb, bias, resid, ldr,
                                         c32, ldc32, c16, ldc16, M, N, K, act);
  };
  auto gemmNN = [&](const _Float16* A, int lda, const _Float16* B, int ldb,
                    const float* bias, const float* resid, int ldr,
                    float* c32, int ldc32, _Float16* c16, int ldc16,
                    int M, int N, int K, int act) {
    dim3 grid((unsigned)(N / 64), (unsigned)(M / 64));
    wk_gemm<0><<<grid, 256, 0, stream>>>(A, lda, B, ldb, bias, resid, ldr,
                                         c32, ldc32, c16, ldc16, M, N, K, act);
  };

  // ---- front end
  wk_sum<<<512, 256, 0, stream>>>(audio, NSAMP, wsum);
  wk_prep<<<g1(NPADLEN), 256, 0, stream>>>(audio, wsum, xp);
  wk_trig<<<g1(NBINS * NFFT), 256, 0, stream>>>(window, tc, ts);
  wk_power<<<NFRAMES, 256, 0, stream>>>(xp, tc, ts, powr);
  wk_mel<<<g1(NMELS * NFRAMES), 256, 0, stream>>>(fbank, powr, mraw, wmax);
  wk_melnorm<<<g1(NMELS * NFRAMES), 256, 0, stream>>>(mraw, wmax, mel);
  wk_conv1<<<g1(DD * NFRAMES), 256, 0, stream>>>(mel, conv1_w, conv1_b, h1);
  wk_conv2<<<g1(DD * TENC), 256, 0, stream>>>(h1, conv2_w, conv2_b, pos_emb, x, xh);

  // ---- encoder layers
  for (int l = 0; l < LEN; ++l) {
    const size_t wlh = (size_t)l * HH * DD * HDIM;
    wk_ln<<<TENC, 128, 0, stream>>>(x, ln1_w + l * DD, ln1_b + l * DD, xn);

    wk_pack_qkv<<<g1(DD * DD), 256, 0, stream>>>(qw + wlh, wB);
    gemmNT(xn, DD, wB, DD, qb + l * DD, nullptr, 0, nullptr, 0, q_h, DD, TP, DD, DD, 0);
    wk_pack_qkv<<<g1(DD * DD), 256, 0, stream>>>(kw + wlh, wB);
    gemmNT(xn, DD, wB, DD, nullptr,     nullptr, 0, nullptr, 0, k_h, DD, TP, DD, DD, 0);
    wk_pack_qkv<<<g1(DD * DD), 256, 0, stream>>>(vw + wlh, wB);
    gemmNT(xn, DD, wB, DD, vb + l * DD, nullptr, 0, nullptr, 0, v_h, DD, TP, DD, DD, 0);

    for (int h = 0; h < HH; ++h) {
      gemmNT(q_h + h * HDIM, DD, k_h + h * HDIM, DD, nullptr, nullptr, 0,
             S, TP, nullptr, 0, TP, TP, HDIM, 0);
      wk_softmax<<<TENC, 256, 0, stream>>>(S, P);
      gemmNN(P, TP, v_h + h * HDIM, DD, nullptr, nullptr, 0,
             nullptr, 0, ao_h + h * HDIM, DD, TP, HDIM, TP, 0);
    }

    wk_pack_o<<<g1(DD * DD), 256, 0, stream>>>(ow + wlh, wB);
    gemmNT(ao_h, DD, wB, DD, ob + l * DD, x, DD, x, DD, nullptr, 0, TP, DD, DD, 0);

    wk_ln<<<TENC, 128, 0, stream>>>(x, ln2_w + l * DD, ln2_b + l * DD, xn);
    wk_pack_nt<<<g1(DD * FFD), 256, 0, stream>>>(fc1_w + (size_t)l * DD * FFD, wB, DD, FFD);
    gemmNT(xn, DD, wB, DD, fc1_b + l * FFD, nullptr, 0, nullptr, 0, ff_h, FFD, TP, FFD, DD, 1);
    wk_pack_nt<<<g1(FFD * DD), 256, 0, stream>>>(fc2_w + (size_t)l * FFD * DD, wB, FFD, DD);
    gemmNT(ff_h, FFD, wB, FFD, fc2_b + l * DD, x, DD, x, DD, nullptr, 0, TP, DD, FFD, 0);
  }

  // ---- final LN + decoder cross-attn caches
  wk_ln<<<TENC, 128, 0, stream>>>(x, lnf_w, lnf_b, xn);
  const size_t keysTotal = (size_t)LEN * HH * HDIM * TENC;   // 4,608,000
  for (int l = 0; l < LEN; ++l) {
    const size_t wlh = (size_t)l * HH * DD * HDIM;
    wk_pack_qkv<<<g1(DD * DD), 256, 0, stream>>>(dec_kw + wlh, wB);
    gemmNT(xn, DD, wB, DD, nullptr, nullptr, 0, Kc, DD, nullptr, 0, TP, DD, DD, 0);
    wk_keys<<<g1(DD * TENC), 256, 0, stream>>>(Kc, out, (size_t)l * DD * TENC);

    wk_pack_qkv<<<g1(DD * DD), 256, 0, stream>>>(dec_vw + wlh, wB);
    gemmNT(xn, DD, wB, DD, dec_vb + l * DD, nullptr, 0, Vc, DD, nullptr, 0, TP, DD, DD, 0);
    wk_vals<<<g1(DD * TENC), 256, 0, stream>>>(Vc, out, keysTotal + (size_t)l * DD * TENC);
  }
}
